// CandidateSampledLossLayer_4861902979674
// MI455X (gfx1250) — compile-verified, matrix operand order checked by reading, not run on hardware
//
#include <hip/hip_runtime.h>
#include <hip/hip_bf16.h>

// ---------------------------------------------------------------------------
// Fused sampled-softmax CE loss for MI455X (gfx1250, wave32, WMMA + TDM).
//   B=16384 rows, D=128, V=1e6, S=8192 sampled, T=1 true label.
// Never materializes the [B,S] logits; online logsumexp per 16-row wave tile.
// GEMM runs as bf16x3 split-precision on v_wmma_f32_16x16x32_bf16.
// Candidate matrix is staged global->LDS via the Tensor Data Mover (double
// buffered, wave-0 issues, s_wait_tensorcnt + barrier), shared by 8 waves.
// ---------------------------------------------------------------------------

typedef __attribute__((ext_vector_type(16))) __bf16        v16bf;
typedef __attribute__((ext_vector_type(8)))  float         v8f;
typedef __attribute__((ext_vector_type(4)))  unsigned int  v4u;
typedef __attribute__((ext_vector_type(8)))  int           v8i;
typedef __attribute__((ext_vector_type(4)))  int           v4i;

#define NUM_SAMPLED 8192
#define TILES_S     (NUM_SAMPLED / 16)   // 512
#define DDIM        128
#define KBLOCKS     (DDIM / 32)          // 4
#define CH          8                    // s-tiles per LDS stage
#define NSTAGES     (TILES_S / CH)       // 64
#define TILE_ELEMS  (KBLOCKS * 32 * 16)  // 2048 bf16 per tile per matrix
#define STAGE_ELEMS (CH * TILE_ELEMS)    // 16384 bf16 (32 KB) per matrix
#define STAGE_BYTES (2 * STAGE_ELEMS * 2)// hi+lo = 64 KB per stage

#if defined(__has_builtin)
#if __has_builtin(__builtin_amdgcn_tensor_load_to_lds) && \
    __has_builtin(__builtin_amdgcn_s_wait_tensorcnt)
#define USE_TDM 1
#endif
#endif

static __device__ __forceinline__ v8f wmma_bf16(v16bf a, v16bf b, v8f c) {
  return __builtin_amdgcn_wmma_f32_16x16x32_bf16(
      false, a, false, b, (short)0, c, false, false);
}

// A/B 16-bit fragment K mapping per CDNA5 ISA 7.12.2 (khalf = lane>>4):
static __device__ __forceinline__ int klocal(int e, int khalf) {
  return (e < 8) ? (e + 8 * khalf) : (e + 8 + 8 * khalf);
}

#ifdef USE_TDM
// 1-D contiguous TDM copy: nelem16 bf16 elements global -> LDS.
// D# packing per CDNA5 ISA §8.3/§8.4 (group0: count/lds/global/type,
// group1: data_size=2B, tensor_dim0=tile_dim0=nelem16, stride=nelem16).
// This toolchain exposes the 6-arg builtin:
//   (uint32x4 g0, int32x8 g1, int32x4 g2, int32x4 g3, int32x8 g4, i32 cpol)
static __device__ __forceinline__ void tdm_load_1d(const void* gptr,
                                                   unsigned lds_addr,
                                                   unsigned nelem16) {
  unsigned long long ga = (unsigned long long)gptr;
  v4u g0;
  g0.x = 1u;                                      // count=1 (valid, user)
  g0.y = lds_addr;                                // lds byte address
  g0.z = (unsigned)(ga & 0xffffffffu);            // global_addr[31:0]
  g0.w = (unsigned)(ga >> 32) | (2u << 30);       // global_addr[56:32]|type=2
  v8i g1;
  g1[0] = 1 << 16;                                // data_size=1 -> 2 bytes
  g1[1] = (int)((nelem16 & 0xffffu) << 16);       // tensor_dim0[15:0]
  g1[2] = (int)((nelem16 >> 16) | (1u << 16));    // tensor_dim0[31:16],dim1=1
  g1[3] = (int)((nelem16 & 0xffffu) << 16);       // tile_dim0
  g1[4] = 0;                                      // tile_dim1/2 unused
  g1[5] = (int)nelem16;                           // tensor_dim0_stride[31:0]
  g1[6] = 0;
  g1[7] = 0;
  v4i gz4 = {0, 0, 0, 0};
  v8i gz8 = {0, 0, 0, 0, 0, 0, 0, 0};
  __builtin_amdgcn_tensor_load_to_lds(g0, g1, gz4, gz4, gz8, 0);
}
#endif

// ---------------------------------------------------------------------------
// Kernel 1: threefry2x32 log-uniform sampler (JAX key(1), n=8192 even split)
// + fused correction corr[s] = bias[s] - log(S) - logQ(s).
// ---------------------------------------------------------------------------
__global__ void sample_kernel(const float* __restrict__ bias,
                              int* __restrict__ sid, float* __restrict__ corr,
                              int V, float log_range, float log_ns) {
  int i = blockIdx.x * blockDim.x + threadIdx.x;   // 0..4095 (pairs)
  if (i >= NUM_SAMPLED / 2) return;

  unsigned x0 = (unsigned)i, x1 = (unsigned)(i + NUM_SAMPLED / 2);
  const unsigned k0 = 0u, k1 = 1u;                 // jax.random.key(1)
  unsigned ks[3] = {k0, k1, 0x1BD11BDAu ^ k0 ^ k1};
  const int R[2][4] = {{13, 15, 26, 6}, {17, 29, 16, 24}};
  x0 += ks[0]; x1 += ks[1];
  #pragma unroll
  for (int g = 0; g < 5; ++g) {
    #pragma unroll
    for (int r = 0; r < 4; ++r) {
      int rot = R[g & 1][r];
      x0 += x1;
      x1 = (x1 << rot) | (x1 >> (32 - rot));
      x1 ^= x0;
    }
    x0 += ks[(g + 1) % 3];
    x1 += ks[(g + 2) % 3] + (unsigned)(g + 1);
  }

  unsigned bits[2] = {x0, x1};
  int idx[2] = {i, i + NUM_SAMPLED / 2};
  #pragma unroll
  for (int p = 0; p < 2; ++p) {
    float u = __uint_as_float(0x3f800000u | (bits[p] >> 9)) - 1.0f;
    int s = (int)floorf(__expf(u * log_range)) - 1;
    s = min(max(s, 0), V - 1);
    sid[idx[p]] = s;
    float q = logf((logf((float)s + 2.0f) - logf((float)s + 1.0f)) / log_range);
    corr[idx[p]] = bias[s] - log_ns - q;
  }
}

// ---------------------------------------------------------------------------
// Kernel 2: gather sampled rows, split fp32 -> bf16 hi/lo, pre-swizzle into
// the WMMA B-matrix per-lane layout (16 contiguous bf16 per lane per frag).
// ---------------------------------------------------------------------------
__global__ void prep_kernel(const float* __restrict__ emb,
                            const int* __restrict__ sid,
                            __bf16* __restrict__ Bhi, __bf16* __restrict__ Blo) {
  int gid = blockIdx.x * blockDim.x + threadIdx.x;   // 512*4*32 = 65536
  int lane = gid & 31;
  int kb   = (gid >> 5) & 3;
  int t    = gid >> 7;
  if (t >= TILES_S) return;

  int n = lane & 15, khalf = lane >> 4;
  int s = sid[t * 16 + n];
  const float* w = emb + (size_t)s * DDIM + kb * 32;
  size_t base = ((size_t)((t * 4 + kb) * 32 + lane)) * 16;
  #pragma unroll
  for (int e = 0; e < 16; ++e) {
    float v = w[klocal(e, khalf)];
    __bf16 h = (__bf16)v;
    Bhi[base + e] = h;
    Blo[base + e] = (__bf16)(v - (float)h);
  }
}

// ---------------------------------------------------------------------------
// Kernel 3: fused GEMM (bf16x3 WMMA, 3 independent acc chains) + corrections
// + accidental-hit mask + branchless online logsumexp + true-logit merge.
// One wave = one 16-row output tile; 8 waves share TDM-staged LDS B tiles.
// ---------------------------------------------------------------------------
__global__ __launch_bounds__(256) void
fused_kernel(const float* __restrict__ movie, const int* __restrict__ target,
             const float* __restrict__ emb, const float* __restrict__ bias,
             const int* __restrict__ sid, const float* __restrict__ corr,
             const __bf16* __restrict__ Bhi, const __bf16* __restrict__ Blo,
             float* __restrict__ out, float log_range, float log_ns) {
  extern __shared__ __align__(128) char smem[];      // 2 * STAGE_BYTES
  int lane  = threadIdx.x & 31;
  int wave  = threadIdx.x >> 5;
  int row0  = (blockIdx.x * 8 + wave) * 16;

  int n     = lane & 15;                      // A-row (M) / B/C-col (N)
  int khalf = lane >> 4;

  // ---- A fragments: 16 rows x 128 dims, bf16 hi/lo, resident in registers
  v16bf a_hi[KBLOCKS], a_lo[KBLOCKS];
  {
    const float* arow = movie + (size_t)(row0 + n) * DDIM;
    #pragma unroll
    for (int kb = 0; kb < KBLOCKS; ++kb) {
      #pragma unroll
      for (int e = 0; e < 16; ++e) {
        float v = arow[kb * 32 + klocal(e, khalf)];
        __bf16 h = (__bf16)v;
        a_hi[kb][e] = h;
        a_lo[kb][e] = (__bf16)(v - (float)h);
      }
    }
  }

  // ---- targets for this lane's 8 accumulator rows (m = j + 8*khalf)
  int tgt[8];
  #pragma unroll
  for (int j = 0; j < 8; ++j) tgt[j] = target[row0 + j + 8 * khalf];

  float rmax[8], rsum[8];
  #pragma unroll
  for (int j = 0; j < 8; ++j) { rmax[j] = -3.0e38f; rsum[j] = 0.0f; }

  const unsigned smem_lds = (unsigned)(size_t)(void*)smem; // LDS byte offset

#ifdef USE_TDM
  // prologue: stage 0 in flight
  if (wave == 0) {
    tdm_load_1d(Bhi, smem_lds, STAGE_ELEMS);
    tdm_load_1d(Blo, smem_lds + STAGE_ELEMS * 2, STAGE_ELEMS);
  }
#endif

  for (int st = 0; st < NSTAGES; ++st) {
    char* buf = smem + (st & 1) * STAGE_BYTES;
#ifdef USE_TDM
    if (wave == 0) {
      if (st + 1 < NSTAGES) {
        unsigned nb = smem_lds + ((st + 1) & 1) * STAGE_BYTES;
        tdm_load_1d(Bhi + (size_t)(st + 1) * STAGE_ELEMS, nb, STAGE_ELEMS);
        tdm_load_1d(Blo + (size_t)(st + 1) * STAGE_ELEMS,
                    nb + STAGE_ELEMS * 2, STAGE_ELEMS);
        __builtin_amdgcn_s_wait_tensorcnt(2);   // stage st complete
      } else {
        __builtin_amdgcn_s_wait_tensorcnt(0);
      }
    }
#else
    {  // fallback: cooperative copy, synchronous (single-buffer semantics)
      const char* gh = (const char*)(Bhi + (size_t)st * STAGE_ELEMS);
      const char* gl = (const char*)(Blo + (size_t)st * STAGE_ELEMS);
      for (int o = threadIdx.x * 16; o < STAGE_ELEMS * 2; o += 256 * 16) {
        *(ulonglong2*)(buf + o) = *(const ulonglong2*)(gh + o);
        *(ulonglong2*)(buf + STAGE_ELEMS * 2 + o) = *(const ulonglong2*)(gl + o);
      }
    }
#endif
    __syncthreads();   // stage st visible (and prior buffer reuse safe)

    const __bf16* lh = (const __bf16*)buf;
    const __bf16* ll = lh + STAGE_ELEMS;

    #pragma unroll 2
    for (int tt = 0; tt < CH; ++tt) {
      int t = st * CH + tt;

      // three independent WMMA accumulation chains (hi*hi, hi*lo, lo*hi)
      v8f acc0 = {}, acc1 = {}, acc2 = {};
      #pragma unroll
      for (int kb = 0; kb < KBLOCKS; ++kb) {
        size_t fo = ((size_t)(tt * 4 + kb) * 32 + lane) * 16;
        v16bf bhv = *(const v16bf*)(lh + fo);
        v16bf blv = *(const v16bf*)(ll + fo);
        acc0 = wmma_bf16(a_hi[kb], bhv, acc0);
        acc1 = wmma_bf16(a_hi[kb], blv, acc1);
        acc2 = wmma_bf16(a_lo[kb], bhv, acc2);
      }

      float cn  = corr[t * 16 + n];
      int   idn = sid[t * 16 + n];
      #pragma unroll
      for (int j = 0; j < 8; ++j) {
        float c = (acc0[j] + acc1[j]) + acc2[j] + cn;
        if (idn == tgt[j]) c -= 1.0e9f;            // accidental-hit mask
        float mnew = fmaxf(rmax[j], c);            // branchless online LSE
        rsum[j] = fmaf(rsum[j], __expf(rmax[j] - mnew), __expf(c - mnew));
        rmax[j] = mnew;
      }
    }
    __syncthreads();   // all waves done with buf before TDM overwrites it
  }

  // ---- butterfly reduce (max,sum) across the 16 lanes of each half
  #pragma unroll
  for (int m = 1; m <= 8; m <<= 1) {
    #pragma unroll
    for (int j = 0; j < 8; ++j) {
      float omx = __shfl_xor(rmax[j], m, 32);
      float osm = __shfl_xor(rsum[j], m, 32);
      float mx  = fmaxf(rmax[j], omx);
      rsum[j] = rsum[j] * __expf(rmax[j] - mx) + osm * __expf(omx - mx);
      rmax[j] = mx;
    }
  }

  // ---- true logit for row (row0 + n): split 128-dim dot across halves
  int trow = row0 + n;
  int tid  = target[trow];
  const float* mrow = movie + (size_t)trow * DDIM + khalf * 64;
  const float* erow = emb + (size_t)tid * DDIM + khalf * 64;
  float dot = 0.0f;
  #pragma unroll
  for (int d = 0; d < 64; d += 4) {
    float4 a = *(const float4*)(mrow + d);
    float4 b = *(const float4*)(erow + d);
    dot = fmaf(a.x, b.x, dot); dot = fmaf(a.y, b.y, dot);
    dot = fmaf(a.z, b.z, dot); dot = fmaf(a.w, b.w, dot);
  }
  dot += __shfl_xor(dot, 16, 32);
  float tl = dot + bias[tid] - log_ns
           - logf((logf((float)tid + 2.0f) - logf((float)tid + 1.0f)) / log_range);

  // ---- merge true logit, write loss (lanes 0-7 -> rows 0-7, 16-23 -> 8-15)
  int   j   = lane & 7;
  int   r   = j + 8 * khalf;
  float tlr = __shfl(tl, r, 32);             // lane r holds row r's true logit
  if ((lane & 15) < 8) {
    float mx = fmaxf(rmax[j], tlr);
    float s  = rsum[j] * __expf(rmax[j] - mx) + __expf(tlr - mx);
    out[row0 + r] = mx + logf(s) - tlr;      // logsumexp - true_logit
  }
}

// ---------------------------------------------------------------------------
extern "C" void kernel_launch(void* const* d_in, const int* in_sizes, int n_in,
                              void* d_out, int out_size, void* d_ws, size_t ws_size,
                              hipStream_t stream) {
  const float* movie  = (const float*)d_in[0];   // [B,128] f32
  const int*   target = (const int*)d_in[1];     // [B,1] int
  const float* emb    = (const float*)d_in[2];   // [V,128] f32
  const float* bias   = (const float*)d_in[3];   // [V] f32

  const int B = in_sizes[0] / DDIM;
  const int V = in_sizes[3];
  const float log_range = logf((float)V + 1.0f);
  const float log_ns    = logf((float)NUM_SAMPLED);

  // workspace layout (hipMalloc'd -> 256B aligned)
  char*   ws   = (char*)d_ws;
  int*    sid  = (int*)ws;                               //  32 KB
  float*  corr = (float*)(ws + 32768);                   //  32 KB
  __bf16* Bhi  = (__bf16*)(ws + 65536);                  //   2 MB
  __bf16* Blo  = (__bf16*)(ws + 65536 + (size_t)NUM_SAMPLED * DDIM * 2);

  sample_kernel<<<(NUM_SAMPLED / 2 + 255) / 256, 256, 0, stream>>>(
      bias, sid, corr, V, log_range, log_ns);
  prep_kernel<<<(TILES_S * 4 * 32 + 255) / 256, 256, 0, stream>>>(
      emb, sid, Bhi, Blo);
  fused_kernel<<<B / 128, 256, 2 * STAGE_BYTES, stream>>>(
      movie, target, emb, bias, sid, corr, Bhi, Blo,
      (float*)d_out, log_range, log_ns);
}
